// MultiHeadedAttention_22428319220354
// MI455X (gfx1250) — matrix operand order repaired and checked by reading, not compiled, not run on hardware
//
#include <hip/hip_runtime.h>

// Problem constants (match reference)
#define DB 4
#define SS 1024
#define DD 1024
#define HH 16
#define HDD 64
#define MROWS (DB * SS) // 4096

typedef __attribute__((ext_vector_type(16))) __bf16 v16bf;
typedef __attribute__((ext_vector_type(8)))  __bf16 v8bf;
typedef __attribute__((ext_vector_type(4)))  __bf16 v4bf;
typedef __attribute__((ext_vector_type(8)))  float  v8f;
typedef unsigned int __attribute__((ext_vector_type(4))) v4u;
typedef int __attribute__((ext_vector_type(4))) v4i;
typedef int __attribute__((ext_vector_type(8))) v8i;

// ---------------------------------------------------------------------------
// Helpers
// ---------------------------------------------------------------------------
__device__ __forceinline__ v8f zero8() {
  v8f z;
#pragma unroll
  for (int r = 0; r < 8; ++r) z[r] = 0.f;
  return z;
}

// bf16 A-fragment (16x32): concat of two contiguous 16B loads -> no repacking.
// Low lanes K = {0..7, 16..23}, high lanes K = {8..15, 24..31}; kbA = lane<16?0:8.
__device__ __forceinline__ v16bf load_a32(const __bf16* p, int kbA) {
  v8bf lo = *(const v8bf*)(p + kbA);
  v8bf hi = *(const v8bf*)(p + 16 + kbA);
  return __builtin_shufflevector(lo, hi, 0, 1, 2, 3, 4, 5, 6, 7,
                                 8, 9, 10, 11, 12, 13, 14, 15);
}

// Issue one TDM 2D tile load: 32 (contiguous K, 2B elems) x 128 rows, row
// stride K, into LDS at ldsAddr. Descriptor per CDNA5 ISA ch.8 (D# groups).
__device__ __forceinline__ void tdm_issue(unsigned ldsAddr, const __bf16* gsrc,
                                          v8i g1, v4i g2, v4i g3) {
  unsigned long long ga = (unsigned long long)(uintptr_t)gsrc;
  v4u g0;
  g0[0] = 1u;                                 // count=1, user mode
  g0[1] = ldsAddr;                            // lds_addr
  g0[2] = (unsigned)ga;                       // global_addr lo32
  g0[3] = (unsigned)(ga >> 32) | (2u << 30);  // global_addr hi, type=2
#if defined(__clang_major__) && (__clang_major__ >= 23)
  v8i zz = {};
  __builtin_amdgcn_tensor_load_to_lds(g0, g1, g2, g3, zz, 0);
#else
  __builtin_amdgcn_tensor_load_to_lds(g0, g1, g2, g3, 0);
#endif
}

// ---------------------------------------------------------------------------
// f32 -> bf16 convert (4 elements / thread)
// ---------------------------------------------------------------------------
__global__ __launch_bounds__(256) void cvt_bf16(const float* __restrict__ src,
                                                __bf16* __restrict__ dst) {
  size_t i = ((size_t)blockIdx.x * 256 + threadIdx.x) * 4;
  float4 v = *(const float4*)(src + i);
  v4bf o;
  o[0] = (__bf16)v.x; o[1] = (__bf16)v.y; o[2] = (__bf16)v.z; o[3] = (__bf16)v.w;
  *(v4bf*)(dst + i) = o;
}

// ---------------------------------------------------------------------------
// W[k][n] f32 -> WT[n][k] bf16  (LDS-tiled 32x32, coalesced both ways)
// ---------------------------------------------------------------------------
__global__ __launch_bounds__(256) void transpose_cvt(const float* __restrict__ W,
                                                     __bf16* __restrict__ WT) {
  __shared__ float tile[32][33];
  int tx = threadIdx.x & 31, ty = threadIdx.x >> 5;
  int nb = blockIdx.x * 32, kb = blockIdx.y * 32;
#pragma unroll
  for (int j = 0; j < 4; ++j)
    tile[ty + 8 * j][tx] = W[(size_t)(kb + ty + 8 * j) * DD + nb + tx];
  __syncthreads();
#pragma unroll
  for (int j = 0; j < 4; ++j)
    WT[(size_t)(nb + ty + 8 * j) * DD + kb + tx] = (__bf16)tile[tx][ty + 8 * j];
}

// ---------------------------------------------------------------------------
// WMMA GEMM: Out = X[M,K] * WT[N,K]^T + bias.  M=4096, N=K=1024.
// Block = 256 threads (8 waves); each wave owns a 16x128 C tile.
// Double-buffered TDM pipeline: wave 0 issues tensor_load_to_lds for tile
// k0+32 into the alternate LDS buffer, then s_wait_tensorcnt(1) for the
// current tile -> DMA of the next tile overlaps this step's 8 WMMAs.
// MODE: 0 = f32 row-major out, 1 = bf16 row-major out,
//       2 = bf16 out transposed per head -> VT[b,h,hd,s] (for the V path).
// ---------------------------------------------------------------------------
template <int MODE>
__global__ __launch_bounds__(256) void wmma_gemm(const __bf16* __restrict__ X,
                                                 const __bf16* __restrict__ WT,
                                                 const float* __restrict__ bias,
                                                 void* __restrict__ outv) {
  __shared__ __bf16 wtile[2][128 * 32]; // 2 x 8KB ping-pong
  const int K = DD, N = DD;
  int lane = threadIdx.x & 31, wid = threadIdx.x >> 5;
  int m0 = (blockIdx.x * 8 + wid) * 16;
  int n0 = blockIdx.y * 128;
  int colq = lane & 15;
  int kbA = (lane < 16) ? 0 : 8;
  int kbB = (lane < 16) ? 0 : 16;
  v8f c[8];
#pragma unroll
  for (int f = 0; f < 8; ++f) c[f] = zero8();

  // Static parts of the TDM descriptor (ISA cdna5 §8)
  v8i g1;
  g1[0] = (1 << 16);                 // workgroup_mask=0, data_size=1 (2 bytes)
  g1[1] = (int)((unsigned)(K & 0xFFFF) << 16);      // tensor_dim0 lo16 @ [63:48]
  g1[2] = (int)(((unsigned)K >> 16) | ((unsigned)(K & 0xFFFF) << 16)); // dim0 hi16, dim1 lo16
  g1[3] = (int)(((unsigned)K >> 16) | (32u << 16)); // dim1 hi16, tile_dim0=32
  g1[4] = 128;                       // tile_dim1=128, tile_dim2=0
  g1[5] = K;                         // tensor_dim0_stride lo32
  g1[6] = 0;                         // stride0 hi16, dim1_stride lo16
  g1[7] = 0;                         // dim1_stride hi32
  v4i g2; g2[0] = 1; g2[1] = 1; g2[2] = 0; g2[3] = 0; // tensor_dim2/3=1, tile_dim3=0
  v4i g3; g3[0] = 0; g3[1] = (int)(1u << 16); g3[2] = 0; g3[3] = 0; // tensor_dim4=1

  unsigned ldsA = (unsigned)(uintptr_t)&wtile[0][0];
  unsigned ldsB = (unsigned)(uintptr_t)&wtile[1][0];
  const __bf16* wsrc = WT + (size_t)n0 * K;
  const __bf16* xrow = X + (size_t)(m0 + colq) * K;

  if (wid == 0) tdm_issue(ldsA, wsrc, g1, g2, g3); // prologue: tile 0

  for (int k0 = 0; k0 < K; k0 += 32) {
    const __bf16* tile = (k0 & 32) ? &wtile[1][0] : &wtile[0][0];
    if (wid == 0) {
      if (k0 + 32 < K) {
        tdm_issue((k0 & 32) ? ldsA : ldsB, wsrc + k0 + 32, g1, g2, g3);
        __builtin_amdgcn_s_wait_tensorcnt(1); // current tile done, next in flight
      } else {
        __builtin_amdgcn_s_wait_tensorcnt(0);
      }
    }
    __syncthreads(); // (a) current tile visible to all waves

    v16bf a = *(const v16bf*)(xrow + k0 + kbB);
#pragma unroll
    for (int f = 0; f < 8; ++f) {
      v16bf bfrag = *(const v16bf*)(tile + (16 * f + colq) * 32 + kbB);
      c[f] = __builtin_amdgcn_wmma_f32_16x16x32_bf16(false, a, false, bfrag,
                                                     (short)0, c[f], false, false);
    }
    __syncthreads(); // (b) tile consumed before it can be overwritten
  }

  int hi8 = (lane >> 4) << 3;
#pragma unroll
  for (int f = 0; f < 8; ++f) {
    int n = n0 + 16 * f + colq;
    float bb = bias[n];
    if (MODE == 2) {
      // V path: write VT[b, h, hd, s]; 8 consecutive s per lane -> 16B store
      int h = n >> 6, hd = n & 63;
      int b = m0 >> 10;
      int s0 = (m0 & (SS - 1)) + hi8;
      v8f acc;
#pragma unroll
      for (int r = 0; r < 8; ++r) acc[r] = c[f][r] + bb;
      v8bf pack = __builtin_convertvector(acc, v8bf);
      __bf16* VT = (__bf16*)outv;
      *(v8bf*)(VT + ((size_t)(b * HH + h) * HDD + hd) * SS + s0) = pack;
    } else {
#pragma unroll
      for (int r = 0; r < 8; ++r) {
        size_t idx = (size_t)(m0 + r + hi8) * N + n;
        float v = c[f][r] + bb;
        if (MODE == 1) ((__bf16*)outv)[idx] = (__bf16)v;
        else           ((float*)outv)[idx]  = v;
      }
    }
  }
}

// ---------------------------------------------------------------------------
// Causal flash attention. One wave per 16-query tile (4 waves / block).
// S^T = K * Q^T so the softmaxed C tile IS the A-fragment of P*V (no
// cross-lane movement). V read from VT[b,h,hd,s] -> contiguous B-fragments.
// launch_bounds(128,1): relax VGPR budget so all 4 K-fragments stay live and
// their loads can clause/overlap instead of load->wait->wmma serialization.
// ---------------------------------------------------------------------------
__global__ __launch_bounds__(128, 1) void attn_fwd(const __bf16* __restrict__ Qb,
                                                   const __bf16* __restrict__ Kb,
                                                   const __bf16* __restrict__ VT,
                                                   const int* __restrict__ maskp,
                                                   __bf16* __restrict__ ctx) {
  int lane = threadIdx.x & 31, wid = threadIdx.x >> 5;
  int t = blockIdx.x;                 // B*H*(S/64) = 1024
  int qb = t & (SS / 64 - 1);
  int bh = t >> 4;
  int h = bh & (HH - 1);
  int b = bh >> 4;
  int q0 = qb * 64 + wid * 16;

  int colq = lane & 15;
  int kbA = (lane < 16) ? 0 : 8;
  int kbB = (lane < 16) ? 0 : 16;
  int hi8 = (lane >> 4) << 3;

  const __bf16* Qp  = Qb + (size_t)b * SS * DD + (size_t)h * HDD;
  const __bf16* Kp  = Kb + (size_t)b * SS * DD + (size_t)h * HDD;
  const __bf16* VTp = VT + (size_t)(b * HH + h) * HDD * SS;
  const int* mp = maskp + b * SS;

  // Q as B-fragments of S^T = K * Q^T (lane column = query index)
  const __bf16* qrow = Qp + (size_t)(q0 + colq) * DD;
  v16bf qf0 = load_a32(qrow, kbA);        // hd 0..31
  v16bf qf1 = load_a32(qrow + 32, kbA);   // hd 32..63
  int q = q0 + colq;

  float mrun = -1e30f, lrun = 0.f;
  v8f o[4];
#pragma unroll
  for (int f = 0; f < 4; ++f) o[f] = zero8();

  const int kmax = q0 + 16;                       // causal: keys 0..q0+15
  for (int k0 = 0; k0 < kmax; k0 += 32) {
    const __bf16* krow0 = Kp + (size_t)(k0 + colq) * DD;
    const __bf16* krow1 = krow0 + (size_t)16 * DD;
    // branchless prefetch of next K block (clamped address, always valid)
    {
      int pfr = k0 + 32 + colq;
      pfr = (pfr < SS) ? pfr : SS - 1;
      __builtin_prefetch(Kp + (size_t)pfr * DD, 0, 1);
    }

    v16bf ka00 = load_a32(krow0, kbA);
    v16bf ka01 = load_a32(krow0 + 32, kbA);
    v16bf ka10 = load_a32(krow1, kbA);
    v16bf ka11 = load_a32(krow1 + 32, kbA);

    v8f s0 = __builtin_amdgcn_wmma_f32_16x16x32_bf16(false, ka01, false, qf1,
                                                     (short)0, zero8(), false, false);
    s0 = __builtin_amdgcn_wmma_f32_16x16x32_bf16(false, ka00, false, qf0,
                                                 (short)0, s0, false, false);
    v8f s1 = __builtin_amdgcn_wmma_f32_16x16x32_bf16(false, ka11, false, qf1,
                                                     (short)0, zero8(), false, false);
    s1 = __builtin_amdgcn_wmma_f32_16x16x32_bf16(false, ka10, false, qf0,
                                                 (short)0, s1, false, false);

    // unconditional vector mask loads (max index = q0+31 <= 1023, in bounds)
    const int* mrow = mp + k0 + hi8;
    int4 ma = *(const int4*)(mrow);
    int4 mb = *(const int4*)(mrow + 4);
    int4 mc = *(const int4*)(mrow + 16);
    int4 md = *(const int4*)(mrow + 20);
    int am[8] = {ma.x, ma.y, ma.z, ma.w, mb.x, mb.y, mb.z, mb.w};
    int bm[8] = {mc.x, mc.y, mc.z, mc.w, md.x, md.y, md.z, md.w};

    // scale 1/sqrt(64)=0.125, causal + input mask (branchless), block row max
    float tmax = -1e30f;
#pragma unroll
    for (int r = 0; r < 8; ++r) {
      int key0 = k0 + r + hi8;
      int key1 = key0 + 16;
      int ok0 = (key0 <= q) & (am[r] != 0);
      int ok1 = (key1 <= q) & (bm[r] != 0);
      float v0 = ok0 ? s0[r] * 0.125f : -1e9f;
      float v1 = ok1 ? s1[r] * 0.125f : -1e9f;
      s0[r] = v0; s1[r] = v1;
      tmax = fmaxf(tmax, fmaxf(v0, v1));
    }
    tmax = fmaxf(tmax, __shfl_xor(tmax, 16, 32));
    float mnew = fmaxf(mrun, tmax);
    float alpha = __expf(mrun - mnew);

    v8f p0v, p1v;
    float rsum = 0.f;
#pragma unroll
    for (int r = 0; r < 8; ++r) {
      p0v[r] = __expf(s0[r] - mnew);
      p1v[r] = __expf(s1[r] - mnew);
      rsum += p0v[r] + p1v[r];
    }
    rsum += __shfl_xor(rsum, 16, 32);
    lrun = lrun * alpha + rsum;
    mrun = mnew;

    // packed f32->bf16: C layout of S^T == A layout of P*V
    v8bf pb0 = __builtin_convertvector(p0v, v8bf);
    v8bf pb1 = __builtin_convertvector(p1v, v8bf);
    v16bf pa = __builtin_shufflevector(pb0, pb1, 0, 1, 2, 3, 4, 5, 6, 7,
                                       8, 9, 10, 11, 12, 13, 14, 15);

    // rescale O rows by alpha[q]; alpha lives in lane == local query index
    float ar[8];
#pragma unroll
    for (int r = 0; r < 8; ++r) ar[r] = __shfl(alpha, r + hi8, 32);
#pragma unroll
    for (int f = 0; f < 4; ++f) {
#pragma unroll
      for (int r = 0; r < 8; ++r) o[f][r] *= ar[r];
    }

    // P[16x32] * V[32x64]: B-fragments are two 16B loads from VT (same K order as pa)
#pragma unroll
    for (int f = 0; f < 4; ++f) {
      v16bf vb = load_a32(VTp + (size_t)(16 * f + colq) * SS + k0, kbA);
      o[f] = __builtin_amdgcn_wmma_f32_16x16x32_bf16(false, pa, false, vb,
                                                     (short)0, o[f], false, false);
    }
  }

  // finalize: divide by l, store context bf16
  float lr[8];
#pragma unroll
  for (int r = 0; r < 8; ++r) {
    float lv = __shfl(lrun, r + hi8, 32);
    lr[r] = (lv > 0.f) ? 1.f / lv : 0.f;
  }
#pragma unroll
  for (int f = 0; f < 4; ++f) {
#pragma unroll
    for (int r = 0; r < 8; ++r) {
      size_t idx = (size_t)((size_t)b * SS + q0 + r + hi8) * DD + h * HDD + 16 * f + colq;
      ctx[idx] = (__bf16)(o[f][r] * lr[r]);
    }
  }
}

// ---------------------------------------------------------------------------
// out = ln_w * (x - mean) / (sqrt(unbiased var) + eps) + ln_b,  x = proj + query
// ---------------------------------------------------------------------------
__global__ __launch_bounds__(256) void ln_residual(const float* __restrict__ proj,
                                                   const float* __restrict__ resid,
                                                   const float* __restrict__ w,
                                                   const float* __restrict__ bparm,
                                                   float* __restrict__ out) {
  __shared__ float redA[8], redB[8];
  int row = blockIdx.x;
  int tid = threadIdx.x;
  int lane = tid & 31, wid = tid >> 5;
  const float* pr = proj + (size_t)row * DD;
  const float* rr = resid + (size_t)row * DD;
  float x[4];
  float s = 0.f;
#pragma unroll
  for (int i = 0; i < 4; ++i) {
    int c = tid + 256 * i;
    x[i] = pr[c] + rr[c];
    s += x[i];
  }
#pragma unroll
  for (int off = 16; off > 0; off >>= 1) s += __shfl_xor(s, off, 32);
  if (lane == 0) redA[wid] = s;
  __syncthreads();
  float tot = 0.f;
#pragma unroll
  for (int j = 0; j < 8; ++j) tot += redA[j];
  float mean = tot * (1.f / DD);
  float vs = 0.f;
#pragma unroll
  for (int i = 0; i < 4; ++i) { float d = x[i] - mean; vs += d * d; }
#pragma unroll
  for (int off = 16; off > 0; off >>= 1) vs += __shfl_xor(vs, off, 32);
  if (lane == 0) redB[wid] = vs;
  __syncthreads();
  float totv = 0.f;
#pragma unroll
  for (int j = 0; j < 8; ++j) totv += redB[j];
  float rstd = 1.f / (sqrtf(totv * (1.f / (DD - 1))) + 1e-6f);
#pragma unroll
  for (int i = 0; i < 4; ++i) {
    int c = tid + 256 * i;
    out[(size_t)row * DD + c] = w[c] * (x[i] - mean) * rstd + bparm[c];
  }
}

// ---------------------------------------------------------------------------
// Host launch
// ---------------------------------------------------------------------------
extern "C" void kernel_launch(void* const* d_in, const int* in_sizes, int n_in,
                              void* d_out, int out_size, void* d_ws, size_t ws_size,
                              hipStream_t stream) {
  (void)in_sizes; (void)n_in; (void)out_size; (void)ws_size;
  const float* query = (const float*)d_in[0];
  const float* keyin = (const float*)d_in[1];
  const float* value = (const float*)d_in[2];
  const int*   mask  = (const int*)d_in[3];
  const float* Wq = (const float*)d_in[4];
  const float* bq = (const float*)d_in[5];
  const float* Wk = (const float*)d_in[6];
  const float* bk = (const float*)d_in[7];
  const float* Wv = (const float*)d_in[8];
  const float* bv = (const float*)d_in[9];
  const float* Wo = (const float*)d_in[10];
  const float* bo = (const float*)d_in[11];
  const float* ln_w = (const float*)d_in[12];
  const float* ln_b = (const float*)d_in[13];
  float* out = (float*)d_out;

  // Workspace layout (~80 MB total)
  char* ws = (char*)d_ws;
  const size_t NE = (size_t)MROWS * DD;            // 4M elements
  const size_t WB = (size_t)DD * DD;               // 1M elements
  __bf16* xq  = (__bf16*)ws; ws += NE * 2;
  __bf16* xk  = (__bf16*)ws; ws += NE * 2;
  __bf16* xv  = (__bf16*)ws; ws += NE * 2;
  __bf16* WqT = (__bf16*)ws; ws += WB * 2;
  __bf16* WkT = (__bf16*)ws; ws += WB * 2;
  __bf16* WvT = (__bf16*)ws; ws += WB * 2;
  __bf16* WoT = (__bf16*)ws; ws += WB * 2;
  __bf16* Qh  = (__bf16*)ws; ws += NE * 2;
  __bf16* Kh  = (__bf16*)ws; ws += NE * 2;
  __bf16* VT  = (__bf16*)ws; ws += NE * 2;         // [B,H,HD,S]
  __bf16* ctx = (__bf16*)ws; ws += NE * 2;
  float*  proj = (float*)ws;

  dim3 cvtGrid((unsigned)(NE / 1024));
  cvt_bf16<<<cvtGrid, 256, 0, stream>>>(query, xq);
  cvt_bf16<<<cvtGrid, 256, 0, stream>>>(keyin, xk);
  cvt_bf16<<<cvtGrid, 256, 0, stream>>>(value, xv);

  dim3 tGrid(DD / 32, DD / 32);
  transpose_cvt<<<tGrid, 256, 0, stream>>>(Wq, WqT);
  transpose_cvt<<<tGrid, 256, 0, stream>>>(Wk, WkT);
  transpose_cvt<<<tGrid, 256, 0, stream>>>(Wv, WvT);
  transpose_cvt<<<tGrid, 256, 0, stream>>>(Wo, WoT);

  dim3 gGrid(MROWS / 128, DD / 128);
  wmma_gemm<1><<<gGrid, 256, 0, stream>>>(xq, WqT, bq, (void*)Qh);
  wmma_gemm<1><<<gGrid, 256, 0, stream>>>(xk, WkT, bk, (void*)Kh);
  wmma_gemm<2><<<gGrid, 256, 0, stream>>>(xv, WvT, bv, (void*)VT);

  attn_fwd<<<DB * HH * (SS / 64), 128, 0, stream>>>(Qh, Kh, VT, mask, ctx);

  wmma_gemm<0><<<gGrid, 256, 0, stream>>>(ctx, WoT, bo, (void*)proj);

  ln_residual<<<MROWS, 256, 0, stream>>>(proj, query, ln_w, ln_b, out);
}
